// SLinOSSMixer_35716948034355
// MI455X (gfx1250) — matrix operand order repaired and checked by reading, not compiled, not run on hardware
//
#include <hip/hip_runtime.h>
#include <hip/hip_bf16.h>
#include <math.h>

typedef __attribute__((ext_vector_type(16))) __bf16 v16bf;
typedef __attribute__((ext_vector_type(8)))  __bf16 v8bf;
typedef __attribute__((ext_vector_type(8)))  float  v8f;

#define D_MODEL 1024
#define D_INNER 2048
#define N_HEADS 32
#define D_HEAD  64
#define D_STATE 128
#define D_CONV  4
#define IN_DIM  (2*D_INNER + N_HEADS + N_HEADS*4*D_STATE)  // 20512
#define BATCH   2
#define SEQ     2048
#define MROWS   (BATCH*SEQ)                                // 4096

#define BM 128
#define BN 64
#define BK 32

__device__ __forceinline__ float sigmoidf_(float x){ return 1.f/(1.f+__expf(-x)); }

// issue one 16-byte async global->LDS copy (ASYNCcnt-tracked)
__device__ __forceinline__ void async_copy_b128(const void* gptr, const void* lptr)
{
  unsigned lds_off = (unsigned)(size_t)lptr;               // LDS aperture: addr[31:0] == LDS byte offset
  unsigned long long gaddr = (unsigned long long)(size_t)gptr;
  asm volatile("global_load_async_to_lds_b128 %0, %1, off"
               :: "v"(lds_off), "v"(gaddr) : "memory");
}

// ---------------------------------------------------------------- cvt f32->bf16
__global__ __launch_bounds__(256) void cvt_f32_bf16(const float* __restrict__ in,
                                                    __bf16* __restrict__ out, int n)
{
  int i = blockIdx.x*256 + threadIdx.x;
  if (i < n) out[i] = (__bf16)in[i];
}

// ---------------------------------------------------------------- WMMA GEMM: C[M,N] = A[M,K] * B[N,K]^T
// A,B bf16 row-major (K contiguous), C f32 row-major (ld = N).
// 128x64 block tile, K-step 32. Double-buffered LDS filled by
// global_load_async_to_lds_b128; pipeline depth 1 via s_wait_asynccnt.
// 8 wave32s; each wave owns 16(M)x64(N) as 4 v_wmma_f32_16x16x32_bf16 accums.
__global__ __launch_bounds__(256,2) void gemm_bt_bf16(
  float* __restrict__ C, const __bf16* __restrict__ A, const __bf16* __restrict__ B,
  int M, int N, int K)
{
  __shared__ __attribute__((aligned(64))) __bf16 As[2][BM][BK];  // 2 x 8 KB
  __shared__ __attribute__((aligned(64))) __bf16 Bs[2][BN][BK];  // 2 x 4 KB

  const int tid  = threadIdx.x;
  const int lane = tid & 31;
  const int wave = tid >> 5;
  const int m0 = blockIdx.y * BM;
  const int n0 = blockIdx.x * BN;

  v8f acc[4] = {};

  // per-thread async-copy assignments (16B granules)
  // A tile: 512 granules -> 2 per thread; B tile: 256 granules -> 1 per thread
  const int a0r = tid >> 2,           a0c = (tid & 3) * 8;          // granule tid
  const int a1r = (tid + 256) >> 2,   a1c = a0c;                    // granule tid+256
  const int br  = tid >> 2,           bc  = (tid & 3) * 8;
  const int brow = (n0 + br < N) ? (n0 + br) : (N - 1);             // clamp OOB (cols masked at store)

  const __bf16* aptr0 = A + (size_t)(m0 + a0r) * K + a0c;
  const __bf16* aptr1 = A + (size_t)(m0 + a1r) * K + a1c;
  const __bf16* bptr  = B + (size_t)brow * K + bc;

  // prologue: fill buffer 0
  async_copy_b128(aptr0, &As[0][a0r][a0c]);
  async_copy_b128(aptr1, &As[0][a1r][a1c]);
  async_copy_b128(bptr,  &Bs[0][br][bc]);

  for (int k0 = 0; k0 < K; k0 += BK) {
    const int cur = (k0 / BK) & 1;
    if (k0 + BK < K) {
      const int nxt = cur ^ 1;
      async_copy_b128(aptr0 + k0 + BK, &As[nxt][a0r][a0c]);
      async_copy_b128(aptr1 + k0 + BK, &As[nxt][a1r][a1c]);
      async_copy_b128(bptr  + k0 + BK, &Bs[nxt][br][bc]);
      asm volatile("s_wait_asynccnt 0x3" ::: "memory");   // current tile's 3 copies done (in-order)
    } else {
      asm volatile("s_wait_asynccnt 0x0" ::: "memory");   // drain for final tile
    }
    __syncthreads();

    // A fragment (16x32 bf16): lane<16 holds K {0..15} of row (lane&15),
    // lane>=16 holds K {16..31}  (contiguous order, matches B)
    const int arow = wave*16 + (lane & 15);
    const int ka   = (lane < 16) ? 0 : 16;
    v8bf alo = *(const v8bf*)&As[cur][arow][ka];
    v8bf ahi = *(const v8bf*)&As[cur][arow][ka + 8];
    v16bf afrag = __builtin_shufflevector(alo, ahi,
        0,1,2,3,4,5,6,7,8,9,10,11,12,13,14,15);

    // B fragment (32x16 bf16): lanes 0-15 -> K=0..15 of col (lane&15), lanes 16-31 -> K=16..31
    const int kb = (lane < 16) ? 0 : 16;
    #pragma unroll
    for (int t = 0; t < 4; t++) {
      v16bf bfrag = *(const v16bf*)&Bs[cur][t*16 + (lane & 15)][kb];
      acc[t] = __builtin_amdgcn_wmma_f32_16x16x32_bf16(
          false, afrag, false, bfrag, (short)0, acc[t], false, false);
    }
    __syncthreads();   // all reads of buf[cur] done before it is refilled next iter
  }

  // C/D layout: VGPR i holds M=i (lanes 0-15) and M=8+i (lanes 16-31), N = lane&15
  const int rbase = m0 + wave*16 + (lane >> 4) * 8;
  const int cb    = n0 + (lane & 15);
  #pragma unroll
  for (int t = 0; t < 4; t++) {
    int cc = cb + t*16;
    if (cc < N) {
      #pragma unroll
      for (int i = 0; i < 8; i++)
        C[(size_t)(rbase + i) * N + cc] = acc[t][i];
    }
  }
}

// ---------------------------------------------------------------- depthwise causal conv(4) + bias + SiLU
__global__ __launch_bounds__(256) void conv_silu_k(
  const float* __restrict__ proj, const float* __restrict__ w,
  const float* __restrict__ bias, float* __restrict__ u)
{
  int idx = blockIdx.x*256 + threadIdx.x;     // over MROWS * D_INNER
  int c   = idx & (D_INNER - 1);
  int row = idx >> 11;                        // D_INNER == 2048
  int l   = row & (SEQ - 1);
  float accv = bias[c];
  #pragma unroll
  for (int j = 0; j < D_CONV; j++) {
    int lp = l - (D_CONV-1) + j;
    if (lp >= 0)
      accv = fmaf(w[c*D_CONV + j],
                  proj[(size_t)(row - (D_CONV-1) + j) * IN_DIM + D_INNER + c], accv);
  }
  u[(size_t)row * D_INNER + c] = accv * sigmoidf_(accv);
}

// ---------------------------------------------------------------- sequential oscillatory scan
// one workgroup (8 wave32s) per (batch, head); y/v states (64x128) in registers,
// per-step operands staged via LDS; writes act = (o + skip*u) * silu(z) as bf16.
__global__ __launch_bounds__(256) void scan_k(
  const float* __restrict__ proj, const float* __restrict__ u,
  const float* __restrict__ dt_bias, const float* __restrict__ theta_raw,
  const float* __restrict__ k_raw, const float* __restrict__ skip,
  __bf16* __restrict__ act)
{
  const int bh = blockIdx.x;           // 0..63
  const int b  = bh >> 5;
  const int h  = bh & (N_HEADS - 1);
  const int tid  = threadIdx.x;
  const int lane = tid & 31;
  const int wave = tid >> 5;
  const int p  = tid >> 2;             // 0..63 (D_HEAD row)
  const int nq = tid & 3;              // quarter of D_STATE

  __shared__ float A_s[D_STATE], kd_s[D_STATE];
  __shared__ float Bn[2*D_STATE], Cn[2*D_STATE];
  __shared__ float u_s[D_HEAD];
  __shared__ float rden[D_STATE];
  __shared__ float redB[8], redC[8];
  __shared__ float opart[4][D_HEAD];
  __shared__ float dt_s;

  if (tid < D_STATE) {
    float th = 3.14159265358979f * sigmoidf_(theta_raw[h*D_STATE + tid]);
    A_s[tid]  = th * th;
    kd_s[tid] = 0.5f * sigmoidf_(k_raw[h*D_STATE + tid]);
  }
  float y[32], v[32];
  #pragma unroll
  for (int i = 0; i < 32; i++) { y[i] = 0.f; v[i] = 0.f; }
  __syncthreads();

  for (int l = 0; l < SEQ; l++) {
    const size_t rowoff = (size_t)(b*SEQ + l);
    const float* prow = proj + rowoff * IN_DIM;
    const float* bcb  = prow + 2*D_INNER + N_HEADS + h*(4*D_STATE);

    float bmv = bcb[tid];                 // Bm: 256 elems == blockDim
    float cmv = bcb[2*D_STATE + tid];     // Cm
    if (tid < D_HEAD) u_s[tid] = u[rowoff*D_INNER + h*D_HEAD + tid];
    if (tid == 0) {
      float raw = prow[2*D_INNER + h] + dt_bias[h];
      float sp  = (raw > 20.f) ? raw : log1pf(__expf(raw));
      dt_s = fminf(fmaxf(sp, 1e-4f), 0.1f);
    }
    // two-level sum-of-squares reduction: wave shfl_xor, then 8 partials in LDS
    float s1 = bmv*bmv, s2 = cmv*cmv;
    #pragma unroll
    for (int off = 16; off > 0; off >>= 1) {
      s1 += __shfl_xor(s1, off, 32);
      s2 += __shfl_xor(s2, off, 32);
    }
    if (lane == 0) { redB[wave] = s1; redC[wave] = s2; }
    __syncthreads();

    float sb = 0.f, sc = 0.f;
    #pragma unroll
    for (int w2 = 0; w2 < 8; w2++) { sb += redB[w2]; sc += redC[w2]; }
    Bn[tid] = bmv * rsqrtf(sb + 1e-8f);
    Cn[tid] = cmv * rsqrtf(sc + 1e-8f);
    float dt = dt_s;
    if (tid < D_STATE)
      rden[tid] = 1.f / (1.f + dt*kd_s[tid] + dt*dt*A_s[tid]);
    __syncthreads();

    float up   = u_s[p];
    float oacc = 0.f;
    const int nb = nq * 32;
    #pragma unroll
    for (int i = 0; i < 32; i++) {
      int n = nb + i;
      float vv = (v[i] - dt*A_s[n]*y[i] + dt*up*Bn[D_STATE + n]) * rden[n];
      float yy = y[i] + dt*vv + dt*up*Bn[n];
      v[i] = vv; y[i] = yy;
      oacc = fmaf(yy, Cn[n], fmaf(vv, Cn[D_STATE + n], oacc));
    }
    opart[nq][p] = oacc;
    __syncthreads();

    if (tid < D_HEAD) {
      int c = h*D_HEAD + tid;
      float o  = opart[0][tid] + opart[1][tid] + opart[2][tid] + opart[3][tid];
      float yi = o + skip[c] * u_s[tid];
      float zv = prow[c];                         // z slice
      act[rowoff*D_INNER + c] = (__bf16)(yi * (zv * sigmoidf_(zv)));
    }
    __syncthreads();
  }
}

// ---------------------------------------------------------------- launch
extern "C" void kernel_launch(void* const* d_in, const int* in_sizes, int n_in,
                              void* d_out, int out_size, void* d_ws, size_t ws_size,
                              hipStream_t stream)
{
  const float* x         = (const float*)d_in[0];
  const float* W_in      = (const float*)d_in[1];
  const float* dw_w      = (const float*)d_in[2];
  const float* dw_b      = (const float*)d_in[3];
  const float* W_out     = (const float*)d_in[4];
  const float* skip      = (const float*)d_in[5];
  const float* dt_bias   = (const float*)d_in[6];
  const float* theta_raw = (const float*)d_in[7];
  const float* k_raw     = (const float*)d_in[8];
  float* out = (float*)d_out;

  char* ws = (char*)d_ws;
  size_t off = 0;
  auto carve = [&](size_t bytes) -> void* {
    void* p = ws + off;
    off += (bytes + 255) & ~(size_t)255;
    return p;
  };
  float*  proj = (float*) carve((size_t)MROWS * IN_DIM  * sizeof(float));   // 336 MB
  float*  uu   = (float*) carve((size_t)MROWS * D_INNER * sizeof(float));   //  32 MB
  __bf16* xb   = (__bf16*)carve((size_t)MROWS * D_MODEL * 2);               //   8 MB
  __bf16* Wib  = (__bf16*)carve((size_t)IN_DIM * D_MODEL * 2);              //  42 MB
  __bf16* Wob  = (__bf16*)carve((size_t)D_MODEL * D_INNER * 2);             //   4 MB
  __bf16* actb = (__bf16*)carve((size_t)MROWS * D_INNER * 2);               //  16 MB
  (void)ws_size; (void)in_sizes; (void)n_in; (void)out_size;

  cvt_f32_bf16<<<(MROWS*D_MODEL + 255)/256, 256, 0, stream>>>(x, xb, MROWS*D_MODEL);
  cvt_f32_bf16<<<(IN_DIM*D_MODEL + 255)/256, 256, 0, stream>>>(W_in, Wib, IN_DIM*D_MODEL);
  cvt_f32_bf16<<<(D_MODEL*D_INNER + 255)/256, 256, 0, stream>>>(W_out, Wob, D_MODEL*D_INNER);

  dim3 g1((IN_DIM + BN - 1)/BN, MROWS/BM);           // 321 x 32
  gemm_bt_bf16<<<g1, 256, 0, stream>>>(proj, xb, Wib, MROWS, IN_DIM, D_MODEL);

  conv_silu_k<<<(MROWS*D_INNER)/256, 256, 0, stream>>>(proj, dw_w, dw_b, uu);

  scan_k<<<BATCH*N_HEADS, 256, 0, stream>>>(proj, uu, dt_bias, theta_raw, k_raw, skip, actb);

  dim3 g2((D_MODEL + BN - 1)/BN, MROWS/BM);          // 16 x 32
  gemm_bt_bf16<<<g2, 256, 0, stream>>>(out, actb, Wob, MROWS, D_MODEL, D_INNER);
}
